// LiteMLA_17892833755325
// MI455X (gfx1250) — matrix operand order repaired and checked by reading, not compiled, not run on hardware
//
#include <hip/hip_runtime.h>
#include <hip/hip_bf16.h>

// ---------------------------------------------------------------------------
// LiteMLA forward for MI455X (gfx1250, wave32, WMMA).
// Memory-bound workload (~2.5GB traffic vs ~60 GFLOP): all dense 1x1-conv
// GEMMs go through v_wmma_f32_16x16x32_f16 with fused epilogues; the dead
// dwc/w3 branch of the reference is skipped entirely.
// Fragments are staged in LDS in WMMA-native layout so each lane builds its
// A/B operands with two ds_load_b128; staging is fully vectorized
// (b64 weight copies, b128 activation loads, packed b32 LDS stores).
// Requires ws_size >= ~1.08 GB.
// ---------------------------------------------------------------------------

typedef __attribute__((ext_vector_type(16))) _Float16 v16h;
typedef __attribute__((ext_vector_type(8)))  _Float16 v8h;
typedef __attribute__((ext_vector_type(4)))  _Float16 v4h;
typedef __attribute__((ext_vector_type(2)))  _Float16 v2h;
typedef __attribute__((ext_vector_type(8)))  float    v8f;

union F16x16 { v16h v; v8h h[2]; _Float16 e[16]; };
union F32x8  { v8f  v; float e[8]; };

#define NTOT 4096      // H*W
#define BATCH 16
#define CMS 1536       // ms channels
#define RSQ8 0.35355339059327373f

__device__ __forceinline__ float sigmoidf_(float x) { return 1.f / (1.f + __expf(-x)); }
__device__ __forceinline__ float softplus_(float x) { return log1pf(__expf(x)); }

// ------------------------------ utility ------------------------------------
__global__ void k_cvt_f16(const float* __restrict__ in, _Float16* __restrict__ out, int n) {
  int i = blockIdx.x * 256 + threadIdx.x;
  if (i < n) out[i] = (_Float16)in[i];
}

__global__ void k_zero(float* __restrict__ p, int n) {
  int i = blockIdx.x * 256 + threadIdx.x;
  if (i < n) p[i] = 0.f;
}

// ------------------------- depthwise conv (SAME) ---------------------------
// grid: (NTOT/256, C, B)
__global__ void k_dwconv(const float* __restrict__ in, const float* __restrict__ w,
                         const float* __restrict__ bias, float* __restrict__ out,
                         int inStrideC, int outStrideC, int K) {
  int s = blockIdx.x * 256 + threadIdx.x;
  int c = blockIdx.y, b = blockIdx.z;
  int y = s >> 6, x = s & 63;
  int pad = (K - 1) >> 1;
  const float* ip = in + ((size_t)b * inStrideC + c) * NTOT;
  const float* wp = w + (size_t)c * K * K;
  float acc = bias ? bias[c] : 0.f;
  for (int ky = 0; ky < K; ++ky) {
    int iy = y + ky - pad;
    if (iy < 0 || iy >= 64) continue;
    for (int kx = 0; kx < K; ++kx) {
      int ix = x + kx - pad;
      if (ix < 0 || ix >= 64) continue;
      acc += wp[ky * K + kx] * ip[iy * 64 + ix];
    }
  }
  out[((size_t)b * outStrideC + c) * NTOT + s] = acc;
}

// ----------------------- grouped 1x1 (agg_pw, groups=96) -------------------
// grid: (NTOT/256, 768, B); out already offset to channel base.
__global__ void k_grouped_pw(const float* __restrict__ in, const float* __restrict__ w,
                             float* __restrict__ out, int inStrideC, int outStrideC) {
  int n = blockIdx.x * 256 + threadIdx.x;
  int o = blockIdx.y, b = blockIdx.z;
  int g = o >> 3;
  const float* ip = in + ((size_t)b * inStrideC + g * 8) * NTOT + n;
  const float* wp = w + (size_t)o * 8;
  float acc = 0.f;
#pragma unroll
  for (int i = 0; i < 8; ++i) acc += wp[i] * ip[(size_t)i * NTOT];
  out[((size_t)b * outStrideC + o) * NTOT + n] = acc;
}

// --------------------------- WMMA GEMM (1x1 conv) --------------------------
// Out[b,o,n] = epilogue( sum_c Wh[o,c] * In[b,c,n] )
// grid: (NTOT/512, O/32, B), block 256 (8 waves).
// Block tile 32(M)x512(N); wave tile 32x64 = 2x4 WMMA accumulators.
// A staged [m][k] (fragment = 2 contiguous 8xf16 runs); B staged transposed
// [n][k] so each B fragment is 2 contiguous 8xf16 runs -> 2x ds_load_b128.
__global__ __launch_bounds__(256) void k_gemm_wmma(
    const _Float16* __restrict__ Wh, const float* __restrict__ In,
    float* __restrict__ Out, int Cdim, int inStrideC, int outStrideC,
    const float* __restrict__ bias,
    const float* __restrict__ bn_g, const float* __restrict__ bn_b,
    const float* __restrict__ bn_m, const float* __restrict__ bn_v,
    const float* __restrict__ addsrc, int addStrideC,
    const float* __restrict__ add_ca, const float* __restrict__ add_wca) {
  __shared__ __align__(16) _Float16 aT[32][40];    // [m][k], 80B row stride
  __shared__ __align__(16) _Float16 bT[512][40];   // [n][k] transposed, 80B row
  int tid  = threadIdx.x;
  int wv   = tid >> 5, lane = tid & 31;
  int half = lane >> 4, lrow = lane & 15;
  int n0 = blockIdx.x * 512;
  int m0 = blockIdx.y * 32;
  int b  = blockIdx.z;

  F32x8 acc[2][4];
#pragma unroll
  for (int mt = 0; mt < 2; ++mt)
#pragma unroll
    for (int nt = 0; nt < 4; ++nt)
      acc[mt][nt].v = (v8f){0.f, 0.f, 0.f, 0.f, 0.f, 0.f, 0.f, 0.f};

  const float* inb = In + (size_t)b * inStrideC * NTOT;

  // staging coordinates (constant per thread)
  const int sa_m  = tid >> 3;          // 0..31
  const int sa_k4 = (tid & 7) << 2;    // 0,4,..,28
  const int sb_n4 = (tid & 127) << 2;  // 0..508 step 4
  const int sb_k0 = tid >> 7;          // 0..1 (base k-pair)

  for (int kk = 0; kk < Cdim; kk += 32) {
    // stage A (weights, already f16): one b64 global load + b64 LDS store
    *(v4h*)&aT[sa_m][sa_k4] =
        *(const v4h*)(Wh + (size_t)(m0 + sa_m) * Cdim + kk + sa_k4);

    // stage B: two-phase. Phase 1: 16 independent b128 loads into regs.
    float4 r0[8], r1[8];
#pragma unroll
    for (int it = 0; it < 8; ++it) {
      int k2 = sb_k0 + it * 2;  // k-pair index 0..15
      r0[it] = *(const float4*)(inb + (size_t)(kk + 2 * k2)     * NTOT + n0 + sb_n4);
      r1[it] = *(const float4*)(inb + (size_t)(kk + 2 * k2 + 1) * NTOT + n0 + sb_n4);
    }
    // Phase 2: convert + packed b32 stores (two f16 per store, adjacent k)
#pragma unroll
    for (int it = 0; it < 8; ++it) {
      int k2 = sb_k0 + it * 2;
      const float* a0 = (const float*)&r0[it];
      const float* a1 = (const float*)&r1[it];
#pragma unroll
      for (int j = 0; j < 4; ++j)
        *(v2h*)&bT[sb_n4 + j][k2 * 2] = (v2h){(_Float16)a0[j], (_Float16)a1[j]};
    }
    if (kk + 32 < Cdim)
      __builtin_prefetch(&inb[(size_t)(kk + 32) * NTOT + n0 + tid], 0, 1);
    __syncthreads();

    // fragments: contiguous 16B chunks -> ds_load_b128 pairs
    F16x16 au[2], bu[4];
#pragma unroll
    for (int mt = 0; mt < 2; ++mt) {
      const _Float16* ar = &aT[mt * 16 + lrow][0];
      au[mt].h[0] = *(const v8h*)(ar + half * 8);        // k = half*8 .. +7
      au[mt].h[1] = *(const v8h*)(ar + 16 + half * 8);   // k = 16+half*8 .. +7
    }
#pragma unroll
    for (int nt = 0; nt < 4; ++nt) {
      const _Float16* br = &bT[wv * 64 + nt * 16 + lrow][0];
      bu[nt].h[0] = *(const v8h*)(br + half * 8);        // k = half*8 .. +7
      bu[nt].h[1] = *(const v8h*)(br + 16 + half * 8);   // k = 16+half*8 .. +7
    }
#pragma unroll
    for (int mt = 0; mt < 2; ++mt)
#pragma unroll
      for (int nt = 0; nt < 4; ++nt)
        acc[mt][nt].v = __builtin_amdgcn_wmma_f32_16x16x32_f16(
            false, au[mt].v, false, bu[nt].v, (short)0, acc[mt][nt].v, false, false);
    __syncthreads();
  }

  // epilogue: D VGPR j -> row m = j + half*8 ; col n = lane&15
#pragma unroll
  for (int mt = 0; mt < 2; ++mt) {
#pragma unroll
    for (int nt = 0; nt < 4; ++nt) {
      int n = n0 + wv * 64 + nt * 16 + lrow;
#pragma unroll
      for (int j = 0; j < 8; ++j) {
        int o = m0 + mt * 16 + half * 8 + j;
        float v = acc[mt][nt].e[j];
        if (bias) v += bias[o];
        if (bn_g) v = (v - bn_m[o]) * rsqrtf(bn_v[o] + 1e-5f) * bn_g[o] + bn_b[o];
        if (addsrc)
          v += addsrc[((size_t)b * addStrideC + o) * NTOT + n] *
               add_ca[(size_t)b * addStrideC + o] * add_wca[b];
        Out[((size_t)b * outStrideC + o) * NTOT + n] = v;
      }
    }
  }
}

// ------------------------ top-k (score = mean+std) -------------------------
// one block per (b,h); bitonic sort 4096 (key = -score) in LDS; keep 128 idx.
__global__ __launch_bounds__(256) void k_topk(const float* __restrict__ ms, int* __restrict__ idx) {
  __shared__ float sk[4096];
  __shared__ int   si[4096];
  int bh = blockIdx.x, b = bh >> 6, h = bh & 63;
  int tid = threadIdx.x;
  const float* qp = ms + ((size_t)b * CMS + h * 24) * NTOT;
  for (int n = tid; n < 4096; n += 256) {
    float q[8], s = 0.f;
#pragma unroll
    for (int d = 0; d < 8; ++d) { q[d] = qp[(size_t)d * NTOT + n]; s += q[d]; }
    float m = s * 0.125f, var = 0.f;
#pragma unroll
    for (int d = 0; d < 8; ++d) { float dd = q[d] - m; var += dd * dd; }
    sk[n] = -(m + sqrtf(var * (1.f / 7.f)));
    si[n] = n;
  }
  for (int k = 2; k <= 4096; k <<= 1) {
    for (int j = k >> 1; j > 0; j >>= 1) {
      __syncthreads();
      for (int i = tid; i < 4096; i += 256) {
        int ixj = i ^ j;
        if (ixj > i) {
          bool up = ((i & k) == 0);
          float a = sk[i], c = sk[ixj];
          if ((a > c) == up) {
            sk[i] = c; sk[ixj] = a;
            int t = si[i]; si[i] = si[ixj]; si[ixj] = t;
          }
        }
      }
    }
  }
  __syncthreads();
  if (tid < 128) idx[bh * 128 + tid] = si[tid];
}

// --------------- sampled attention row (softmax + keep-top-64) -------------
// grid (128, 64, B), block 128: one selected query row per block.
__global__ __launch_bounds__(128) void k_sampled(const float* __restrict__ ms,
                                                 const int* __restrict__ idx,
                                                 float* __restrict__ scat) {
  __shared__ int   sidx[128];
  __shared__ float qrow[8];
  __shared__ float wsh[128];
  __shared__ float red[128];
  __shared__ float part[128][8];
  int t = blockIdx.x, h = blockIdx.y, b = blockIdx.z;
  int s = threadIdx.x;
  int bh = b * 64 + h;
  sidx[s] = idx[bh * 128 + s];
  __syncthreads();
  int nt = sidx[t];
  const float* base = ms + ((size_t)b * CMS + h * 24) * NTOT;
  if (s < 8) qrow[s] = base[(size_t)s * NTOT + nt];
  __syncthreads();
  int ns = sidx[s];
  float lg = 0.f;
#pragma unroll
  for (int d = 0; d < 8; ++d) lg += qrow[d] * base[(size_t)(8 + d) * NTOT + ns];
  lg *= RSQ8;
  red[s] = lg; __syncthreads();
  for (int off = 64; off > 0; off >>= 1) { if (s < off) red[s] = fmaxf(red[s], red[s + off]); __syncthreads(); }
  float mx = red[0]; __syncthreads();
  float e = __expf(lg - mx);
  red[s] = e; __syncthreads();
  for (int off = 64; off > 0; off >>= 1) { if (s < off) red[s] += red[s + off]; __syncthreads(); }
  float w = e / red[0];
  __syncthreads();
  wsh[s] = w;
  for (int k = 2; k <= 128; k <<= 1) {
    for (int j = k >> 1; j > 0; j >>= 1) {
      __syncthreads();
      int i = s, ixj = s ^ j;
      if (ixj > i) {
        bool up = ((i & k) == 0);
        float a = wsh[i], c = wsh[ixj];
        if ((a > c) == up) { wsh[i] = c; wsh[ixj] = a; }
      }
    }
  }
  __syncthreads();
  float thr = wsh[64];            // ascending sort: index -64 of 128
  float wk = (w >= thr) ? w : 0.f;
#pragma unroll
  for (int d = 0; d < 8; ++d) part[s][d] = wk * base[(size_t)(16 + d) * NTOT + ns];
  __syncthreads();
  if (s < 8) {
    float acc = 0.f;
    for (int i = 0; i < 128; ++i) acc += part[i][s];
    scat[((size_t)bh * 4096 + nt) * 8 + s] = acc;
  }
}

// ------------- q/k column-norm factors: ||q||_N / ||q^3||_N per (b,h,d) ----
__global__ __launch_bounds__(256) void k_qknorm(const float* __restrict__ ms,
                                                const float* __restrict__ scale_p,
                                                float* __restrict__ qfac, float* __restrict__ kfac) {
  __shared__ float red[256][32];
  int bh = blockIdx.x, b = bh >> 6, h = bh & 63;
  int tid = threadIdx.x;
  float inv[8];
#pragma unroll
  for (int d = 0; d < 8; ++d) inv[d] = 1.f / softplus_(scale_p[d]);
  const float* qp = ms + ((size_t)b * CMS + h * 24) * NTOT;
  const float* kp = qp + (size_t)8 * NTOT;
  float a[32];
#pragma unroll
  for (int i = 0; i < 32; ++i) a[i] = 0.f;
  for (int n = tid; n < 4096; n += 256) {
#pragma unroll
    for (int d = 0; d < 8; ++d) {
      float q = qp[(size_t)d * NTOT + n]; q = (q > 0.f ? q : 0.f) + 1e-6f; q *= inv[d];
      float q2 = q * q; a[d] += q2; a[8 + d] += q2 * q2 * q2;
      float kx = kp[(size_t)d * NTOT + n]; kx = (kx > 0.f ? kx : 0.f) + 1e-6f; kx *= inv[d];
      float k2 = kx * kx; a[16 + d] += k2; a[24 + d] += k2 * k2 * k2;
    }
  }
#pragma unroll
  for (int i = 0; i < 32; ++i) red[tid][i] = a[i];
  __syncthreads();
  for (int off = 128; off > 0; off >>= 1) {
    if (tid < off)
      for (int i = 0; i < 32; ++i) red[tid][i] += red[tid + off][i];
    __syncthreads();
  }
  if (tid < 8) {
    qfac[bh * 8 + tid] = sqrtf(red[0][tid])      / sqrtf(red[0][8 + tid]);
    kfac[bh * 8 + tid] = sqrtf(red[0][16 + tid]) / sqrtf(red[0][24 + tid]);
  }
}

// ----------------- kv[d][e] = kfac[d] * sum_n k^3 * vpad ; grid (9, B*64) --
__global__ __launch_bounds__(256) void k_kv(const float* __restrict__ ms,
                                            const float* __restrict__ scale_p,
                                            const float* __restrict__ kfac,
                                            float* __restrict__ kv) {
  __shared__ float red[256];
  int e = blockIdx.x;
  int bh = blockIdx.y, b = bh >> 6, h = bh & 63;
  int tid = threadIdx.x;
  float inv[8];
#pragma unroll
  for (int d = 0; d < 8; ++d) inv[d] = 1.f / softplus_(scale_p[d]);
  const float* kp = ms + ((size_t)b * CMS + h * 24 + 8) * NTOT;
  const float* vp = ms + ((size_t)b * CMS + h * 24 + 16) * NTOT;
  float acc[8];
#pragma unroll
  for (int d = 0; d < 8; ++d) acc[d] = 0.f;
  for (int n = tid; n < 4096; n += 256) {
    float v = (e < 8) ? vp[(size_t)e * NTOT + n] : 1.f;
#pragma unroll
    for (int d = 0; d < 8; ++d) {
      float kx = kp[(size_t)d * NTOT + n]; kx = (kx > 0.f ? kx : 0.f) + 1e-6f; kx *= inv[d];
      acc[d] += kx * kx * kx * v;
    }
  }
  for (int d = 0; d < 8; ++d) {
    red[tid] = acc[d]; __syncthreads();
    for (int off = 128; off > 0; off >>= 1) { if (tid < off) red[tid] += red[tid + off]; __syncthreads(); }
    if (tid == 0) kv[(size_t)bh * 72 + d * 9 + e] = kfac[bh * 8 + d] * red[0];
    __syncthreads();
  }
}

// ---------------- out[n,e] = (q^3*qfac) @ kv ; /denominator --------------
__global__ __launch_bounds__(256) void k_attnout(const float* __restrict__ ms,
                                                 const float* __restrict__ scale_p,
                                                 const float* __restrict__ qfac,
                                                 const float* __restrict__ kvbuf,
                                                 float* __restrict__ lin) {
  __shared__ float kv[72];
  __shared__ float qf[8];
  int bh = blockIdx.y, b = bh >> 6, h = bh & 63;
  int tid = threadIdx.x;
  int n = blockIdx.x * 256 + tid;
  if (tid < 72) kv[tid] = kvbuf[(size_t)bh * 72 + tid];
  if (tid < 8)  qf[tid] = qfac[bh * 8 + tid];
  __syncthreads();
  float inv[8];
#pragma unroll
  for (int d = 0; d < 8; ++d) inv[d] = 1.f / softplus_(scale_p[d]);
  const float* qp = ms + ((size_t)b * CMS + h * 24) * NTOT;
  float qfin[8];
#pragma unroll
  for (int d = 0; d < 8; ++d) {
    float q = qp[(size_t)d * NTOT + n]; q = (q > 0.f ? q : 0.f) + 1e-6f; q *= inv[d];
    qfin[d] = q * q * q * qf[d];
  }
  float den = 1e-15f;
#pragma unroll
  for (int d = 0; d < 8; ++d) den += qfin[d] * kv[d * 9 + 8];
  float invden = 1.f / den;
#pragma unroll
  for (int e = 0; e < 8; ++e) {
    float num = 0.f;
#pragma unroll
    for (int d = 0; d < 8; ++d) num += qfin[d] * kv[d * 9 + e];
    lin[((size_t)b * 512 + h * 8 + e) * NTOT + n] = num * invden;
  }
}

// ------------------------------- pooling -----------------------------------
__global__ __launch_bounds__(256) void k_pool(const float* __restrict__ in, float* __restrict__ out, int strideC) {
  __shared__ float red[256];
  int c = blockIdx.x, b = blockIdx.y, tid = threadIdx.x;
  const float* p = in + ((size_t)b * strideC + c) * NTOT;
  float a = 0.f;
  for (int n = tid; n < 4096; n += 256) a += p[n];
  red[tid] = a; __syncthreads();
  for (int off = 128; off > 0; off >>= 1) { if (tid < off) red[tid] += red[tid + off]; __syncthreads(); }
  if (tid == 0) out[(size_t)b * gridDim.x + c] = red[0] * (1.f / 4096.f);
}

// pooled_sample: mean over the interleaved view of scat (c -> (h, n-range, d))
__global__ __launch_bounds__(256) void k_pool_scat(const float* __restrict__ scat, float* __restrict__ out) {
  __shared__ float red[256];
  int c = blockIdx.x, b = blockIdx.y, tid = threadIdx.x;
  int h = c >> 3, nbase = (c & 7) * 512;
  const float* p = scat + (((size_t)(b * 64 + h)) * 4096 + nbase) * 8;   // contiguous 4096 floats
  float a = 0.f;
  for (int i = tid; i < 4096; i += 256) a += p[i];
  red[tid] = a; __syncthreads();
  for (int off = 128; off > 0; off >>= 1) { if (tid < off) red[tid] += red[tid + off]; __syncthreads(); }
  if (tid == 0) out[(size_t)b * 512 + c] = red[0] * (1.f / 4096.f);
}

// ----------------------- channel attention (rc path) -----------------------
__global__ __launch_bounds__(256) void k_ca(const float* __restrict__ p,
    const float* __restrict__ w1, const float* __restrict__ b1,
    const float* __restrict__ w2, const float* __restrict__ b2,
    const float* __restrict__ wgo_w, const float* __restrict__ wgo_b,
    float* __restrict__ ca, float* __restrict__ wca) {
  __shared__ float ps[256];
  __shared__ float hid[16];
  __shared__ float red[256];
  int b = blockIdx.x, tid = threadIdx.x;
  ps[tid] = p[(size_t)b * 256 + tid];
  __syncthreads();
  if (tid < 16) {
    float a = b1[tid];
    for (int c = 0; c < 256; ++c) a += w1[tid * 256 + c] * ps[c];
    hid[tid] = a > 0.f ? a : 0.f;
  }
  __syncthreads();
  float a = b2[tid];
#pragma unroll
  for (int hh = 0; hh < 16; ++hh) a += w2[tid * 16 + hh] * hid[hh];
  float cav = sigmoidf_(a);
  ca[(size_t)b * 256 + tid] = cav;
  red[tid] = ps[tid] * cav * wgo_w[tid];
  __syncthreads();
  for (int off = 128; off > 0; off >>= 1) { if (tid < off) red[tid] += red[tid + off]; __syncthreads(); }
  if (tid == 0) wca[b] = sigmoidf_(red[0] + wgo_b[0]);
}

// ------------------------ gating weights w1,w2 -----------------------------
__global__ __launch_bounds__(256) void k_wg(const float* __restrict__ psamp, const float* __restrict__ pfoc,
                                            const float* __restrict__ wg_w, const float* __restrict__ wg_b,
                                            float* __restrict__ w12) {
  __shared__ float r1[256], r2[256];
  int b = blockIdx.x, tid = threadIdx.x;
  float a1 = 0.f, a2 = 0.f;
  for (int j = tid; j < 512; j += 256) {
    a1 += wg_w[j]       * psamp[(size_t)b * 512 + j];
    a2 += wg_w[512 + j] * pfoc[(size_t)b * 512 + j];
  }
  r1[tid] = a1; r2[tid] = a2; __syncthreads();
  for (int off = 128; off > 0; off >>= 1) {
    if (tid < off) { r1[tid] += r1[tid + off]; r2[tid] += r2[tid + off]; }
    __syncthreads();
  }
  if (tid == 0) { w12[b * 2 + 0] = sigmoidf_(r1[0] + wg_b[0]); w12[b * 2 + 1] = sigmoidf_(r2[0] + wg_b[1]); }
}

// ---------------- combine: lin*w2 + sampled_result*w1 ----------------------
__global__ void k_combine(const float* __restrict__ lin, const float* __restrict__ scat,
                          const float* __restrict__ w12, float* __restrict__ out) {
  int s = blockIdx.x * 256 + threadIdx.x;
  int c = blockIdx.y, b = blockIdx.z;
  float w1 = w12[b * 2 + 0], w2 = w12[b * 2 + 1];
  int h = c >> 3;
  int r = ((c & 7) << 12) + s;            // interleaved reshape mapping of v/scat
  int n = r >> 3, d = r & 7;
  float samp = scat[(((size_t)(b * 64 + h)) * 4096 + n) * 8 + d];
  float l = lin[((size_t)b * 512 + c) * NTOT + s];
  out[((size_t)b * 512 + c) * NTOT + s] = l * w2 + samp * w1;
}

// ---------------------------------------------------------------------------
extern "C" void kernel_launch(void* const* d_in, const int* in_sizes, int n_in,
                              void* d_out, int out_size, void* d_ws, size_t ws_size,
                              hipStream_t stream) {
  (void)in_sizes; (void)n_in; (void)out_size; (void)ws_size;
  const float* x        = (const float*)d_in[0];
  const float* qkv_w    = (const float*)d_in[1];
  const float* agg_dw_w = (const float*)d_in[2];
  const float* agg_pw_w = (const float*)d_in[3];
  const float* proj_w   = (const float*)d_in[4];
  const float* bn_g     = (const float*)d_in[5];
  const float* bn_b     = (const float*)d_in[6];
  const float* bn_m     = (const float*)d_in[7];
  const float* bn_v     = (const float*)d_in[8];
  // d_in[9]/d_in[10] (dwc_w, dwc_b) are dead in the reference (w3 unused).
  const float* scale_p  = (const float*)d_in[11];
  const float* c3_dw_w  = (const float*)d_in[12];
  const float* c3_dw_b  = (const float*)d_in[13];
  const float* c3_pw_w  = (const float*)d_in[14];
  const float* c3_pw_b  = (const float*)d_in[15];
  const float* c5_dw_w  = (const float*)d_in[16];
  const float* c5_dw_b  = (const float*)d_in[17];
  const float* c5_pw_w  = (const float*)d_in[18];
  const float* c5_pw_b  = (const float*)d_in[19];
  const float* ca_w1    = (const float*)d_in[20];
  const float* ca_b1    = (const float*)d_in[21];
  const float* ca_w2    = (const float*)d_in[22];
  const float* ca_b2    = (const float*)d_in[23];
  const float* wg_w     = (const float*)d_in[24];
  const float* wg_b     = (const float*)d_in[25];
  const float* wgo_w    = (const float*)d_in[26];
  const float* wgo_b    = (const float*)d_in[27];

  size_t off = 0;
  char* base = (char*)d_ws;
  auto alloc = [&](size_t bytes) -> void* {
    void* p = (void*)(base + off);
    off += (bytes + 255) & ~(size_t)255;
    return p;
  };
  float* ms    = (float*)alloc((size_t)BATCH * CMS * NTOT * 4);   // 384 MiB
  float* tmpa  = (float*)alloc((size_t)BATCH * 768 * NTOT * 4);   // 192 MiB (reused as comb)
  float* t1    = (float*)alloc((size_t)BATCH * 256 * NTOT * 4);
  float* t2    = (float*)alloc((size_t)BATCH * 256 * NTOT * 4);
  float* rc    = (float*)alloc((size_t)BATCH * 256 * NTOT * 4);
  float* scat  = (float*)alloc((size_t)BATCH * 64 * NTOT * 8 * 4);
  float* lin   = (float*)alloc((size_t)BATCH * 512 * NTOT * 4);
  int*   idx   = (int*)  alloc((size_t)BATCH * 64 * 128 * 4);
  float* qfac  = (float*)alloc((size_t)BATCH * 64 * 8 * 4);
  float* kfac  = (float*)alloc((size_t)BATCH * 64 * 8 * 4);
  float* kvbuf = (float*)alloc((size_t)BATCH * 64 * 72 * 4);
  float* pbuf  = (float*)alloc((size_t)BATCH * 256 * 4);
  float* cabuf = (float*)alloc((size_t)BATCH * 256 * 4);
  float* wca   = (float*)alloc((size_t)BATCH * 4);
  float* pfoc  = (float*)alloc((size_t)BATCH * 512 * 4);
  float* psmp  = (float*)alloc((size_t)BATCH * 512 * 4);
  float* w12   = (float*)alloc((size_t)BATCH * 2 * 4);
  _Float16* qkv_h  = (_Float16*)alloc((size_t)768 * 256 * 2);
  _Float16* c3_h   = (_Float16*)alloc((size_t)256 * 256 * 2);
  _Float16* c5_h   = (_Float16*)alloc((size_t)256 * 256 * 2);
  _Float16* proj_h = (_Float16*)alloc((size_t)256 * 512 * 2);
  float* comb = tmpa;   // reuse: agg dw scratch dead after grouped_pw

  // weight precision conversion (f32 -> f16 for WMMA A operand)
  k_cvt_f16<<<(768 * 256 + 255) / 256, 256, 0, stream>>>(qkv_w, qkv_h, 768 * 256);
  k_cvt_f16<<<(256 * 256 + 255) / 256, 256, 0, stream>>>(c3_pw_w, c3_h, 256 * 256);
  k_cvt_f16<<<(256 * 256 + 255) / 256, 256, 0, stream>>>(c5_pw_w, c5_h, 256 * 256);
  k_cvt_f16<<<(256 * 512 + 255) / 256, 256, 0, stream>>>(proj_w, proj_h, 256 * 512);

  // ---- rc path: dw3 -> pw(WMMA) -> dw5 -> pw(WMMA) -> pool -> CA ----
  k_dwconv<<<dim3(16, 256, BATCH), 256, 0, stream>>>(x, c3_dw_w, c3_dw_b, t1, 256, 256, 3);
  k_gemm_wmma<<<dim3(8, 8, BATCH), 256, 0, stream>>>(c3_h, t1, t2, 256, 256, 256, c3_pw_b,
      nullptr, nullptr, nullptr, nullptr, nullptr, 0, nullptr, nullptr);
  k_dwconv<<<dim3(16, 256, BATCH), 256, 0, stream>>>(t2, c5_dw_w, c5_dw_b, t1, 256, 256, 5);
  k_gemm_wmma<<<dim3(8, 8, BATCH), 256, 0, stream>>>(c5_h, t1, rc, 256, 256, 256, c5_pw_b,
      nullptr, nullptr, nullptr, nullptr, nullptr, 0, nullptr, nullptr);
  k_pool<<<dim3(256, BATCH), 256, 0, stream>>>(rc, pbuf, 256);
  k_ca<<<BATCH, 256, 0, stream>>>(pbuf, ca_w1, ca_b1, ca_w2, ca_b2, wgo_w, wgo_b, cabuf, wca);

  // ---- qkv (WMMA) + agg (dw5 + grouped pw) into ms ----
  k_gemm_wmma<<<dim3(8, 24, BATCH), 256, 0, stream>>>(qkv_h, x, ms, 256, 256, CMS, nullptr,
      nullptr, nullptr, nullptr, nullptr, nullptr, 0, nullptr, nullptr);
  k_dwconv<<<dim3(16, 768, BATCH), 256, 0, stream>>>(ms, agg_dw_w, nullptr, tmpa, CMS, 768, 5);
  k_grouped_pw<<<dim3(16, 768, BATCH), 256, 0, stream>>>(tmpa, agg_pw_w, ms + (size_t)768 * NTOT, 768, CMS);

  // ---- sparse sampled attention ----
  k_topk<<<BATCH * 64, 256, 0, stream>>>(ms, idx);
  k_zero<<<(BATCH * 64 * NTOT * 8 + 255) / 256, 256, 0, stream>>>(scat, BATCH * 64 * NTOT * 8);
  k_sampled<<<dim3(128, 64, BATCH), 128, 0, stream>>>(ms, idx, scat);

  // ---- relu linear attention ----
  k_qknorm<<<BATCH * 64, 256, 0, stream>>>(ms, scale_p, qfac, kfac);
  k_kv<<<dim3(9, BATCH * 64), 256, 0, stream>>>(ms, scale_p, kfac, kvbuf);
  k_attnout<<<dim3(16, BATCH * 64), 256, 0, stream>>>(ms, scale_p, qfac, kvbuf, lin);

  // ---- gating + combine ----
  k_pool<<<dim3(512, BATCH), 256, 0, stream>>>(lin, pfoc, 512);
  k_pool_scat<<<dim3(512, BATCH), 256, 0, stream>>>(scat, psmp);
  k_wg<<<BATCH, 256, 0, stream>>>(psmp, pfoc, wg_w, wg_b, w12);
  k_combine<<<dim3(16, 512, BATCH), 256, 0, stream>>>(lin, scat, w12, comb);

  // ---- proj (WMMA) with fused BN + rc*ca*w_ca residual -> d_out ----
  k_gemm_wmma<<<dim3(8, 8, BATCH), 256, 0, stream>>>(proj_h, comb, (float*)d_out, 512, 512, 256,
      nullptr, bn_g, bn_b, bn_m, bn_v, rc, 256, cabuf, wca);
}
